// LSTM_30107720745581
// MI455X (gfx1250) — matrix operand order repaired and checked by reading, not compiled, not run on hardware
//
#include <hip/hip_runtime.h>

// ---------------- types ----------------
typedef __attribute__((ext_vector_type(16))) __bf16 v16bf;
typedef __attribute__((ext_vector_type(8)))  __bf16 v8bf;
typedef __attribute__((ext_vector_type(8)))  float  v8f;

// problem constants
constexpr int kT  = 512;
constexpr int kB  = 32;
constexpr int kH  = 1024;
constexpr int kG  = 4096;          // 4*H
constexpr int kCIN = 1024;
constexpr int kBH = kB * kH;       // 32768
constexpr int kB4H = kB * kG;      // 131072
constexpr int kTB = kT * kB;       // 16384

// LDS layout strides (padded to avoid bank conflicts on 64x4B banks)
constexpr int kLdsBStride = 72;    // 64 cols + 8 pad -> 144B row stride
constexpr int kLdsAStride = 1032;  // 1024 + 8 pad    -> 2064B row stride

// ---------------- wmma helpers ----------------
__device__ __forceinline__ v8f bwmma(v16bf a, v16bf b, v8f c) {
  // D = A(16x32 bf16) * B(32x16 bf16) + C(16x16 f32)
  return __builtin_amdgcn_wmma_f32_16x16x32_bf16(
      /*neg_a=*/false, a, /*neg_b=*/false, b,
      /*c_mod=*/(short)0, c, /*reuse_a=*/false, /*reuse_b=*/false);
}

// A-fragment (16x32), row-major source (global or LDS), lda in elements.
// ISA 7.12.2: lanes 0-15 -> M=lane, K 0..7 & 16..23; lanes 16-31 -> M=lane-16, K 8..15 & 24..31.
__device__ __forceinline__ v16bf load_a_frag(const __bf16* A, int lda,
                                             int m0, int k0, int lane) {
  union { v16bf v; v8bf h[2]; } u;
  const __bf16* base = A + (size_t)(m0 + (lane & 15)) * lda + k0 + ((lane >> 4) << 3);
  u.h[0] = *(const v8bf*)(base);
  u.h[1] = *(const v8bf*)(base + 16);
  return u.v;
}

// B-fragment (32x16), row-major K x N source, lane%16 = N col; lanes 0-15 K 0..7 & 16..23, lanes 16-31 K 8..15 & 24..31.
__device__ __forceinline__ v16bf load_b_frag(const __bf16* Bm, int ldb,
                                             int k0, int n0, int lane) {
  union { v16bf v; v8bf h[2]; } u;
  const __bf16* p = Bm + (size_t)(k0 + ((lane >> 4) << 3)) * ldb + n0 + (lane & 15);
#pragma unroll
  for (int s = 0; s < 8; ++s) {
    u.h[0][s] = p[(size_t)s * ldb];
    u.h[1][s] = p[(size_t)(16 + s) * ldb];
  }
  return u.v;
}

// CDNA5 async DMA: memory -> LDS, 16 bytes per lane, tracked by ASYNCcnt.
__device__ __forceinline__ void async_ld_lds_b128(void* lds_dst, const void* gsrc) {
  unsigned laddr = (unsigned)(unsigned long long)lds_dst;      // low 32 bits = LDS offset
  unsigned long long gaddr = (unsigned long long)gsrc;
  asm volatile("global_load_async_to_lds_b128 %0, %1, off"
               :: "v"(laddr), "v"(gaddr) : "memory");
}
__device__ __forceinline__ void wait_asynccnt0() {
  asm volatile("s_wait_asynccnt 0" ::: "memory");
}

// ---------------- small utility kernels ----------------
__global__ void zero_u32(unsigned* p, int n) {
  int i = blockIdx.x * blockDim.x + threadIdx.x;
  if (i < n) p[i] = 0u;
}

__global__ void cvt_f32_bf16(const float* __restrict__ s, __bf16* __restrict__ d, long n) {
  long i = (long)blockIdx.x * blockDim.x + threadIdx.x;
  long stride = (long)gridDim.x * blockDim.x;
  for (; i < n; i += stride) d[i] = (__bf16)s[i];
}

// ---------------- GEMM: C(MxN,f32) = A(MxK,bf16) @ B(KxN,bf16) + bias(N) ----------------
// block = 256 threads = 8 waves arranged 2(M) x 4(N); wave tile = 32x64; block tile = 64x256.
__global__ __launch_bounds__(256) void gemm_bias(const __bf16* __restrict__ A,
                                                 const __bf16* __restrict__ Bm,
                                                 const float* __restrict__ bias,
                                                 float* __restrict__ C,
                                                 int M, int K, int N) {
  const int lane = threadIdx.x & 31;
  const int wave = threadIdx.x >> 5;
  const int m0 = blockIdx.x * 64 + (wave >> 2) * 32;
  const int n0 = blockIdx.y * 256 + (wave & 3) * 64;

  const v8f vzero = {0.f, 0.f, 0.f, 0.f, 0.f, 0.f, 0.f, 0.f};
  v8f acc[2][4];
#pragma unroll
  for (int i = 0; i < 2; ++i)
#pragma unroll
    for (int j = 0; j < 4; ++j) acc[i][j] = vzero;

  for (int k0 = 0; k0 < K; k0 += 32) {
    v16bf a0 = load_a_frag(A, K, m0, k0, lane);
    v16bf a1 = load_a_frag(A, K, m0 + 16, k0, lane);
#pragma unroll
    for (int j = 0; j < 4; ++j) {
      v16bf b = load_b_frag(Bm, N, k0, n0 + j * 16, lane);
      acc[0][j] = bwmma(a0, b, acc[0][j]);
      acc[1][j] = bwmma(a1, b, acc[1][j]);
    }
  }

  const int coll = lane & 15;
  const int rsel = (lane >> 4) * 8;
#pragma unroll
  for (int i = 0; i < 2; ++i) {
    const int rbase = m0 + i * 16 + rsel;
#pragma unroll
    for (int j = 0; j < 4; ++j) {
      const int col = n0 + j * 16 + coll;
      const float bv = bias[col];
#pragma unroll
      for (int r = 0; r < 8; ++r)
        C[(size_t)(rbase + r) * N + col] = acc[i][j][r] + bv;
    }
  }
}

// ---------------- grid-wide barrier (monotonic generation) ----------------
__device__ __forceinline__ void grid_barrier(int* cnt, int* gen, int nwg, int it) {
  __threadfence();     // make this WG's h-tile stores visible device-wide
  __syncthreads();
  if (threadIdx.x == 0) {
    int old = atomicAdd(cnt, 1);
    if (old == nwg * (it + 1) - 1) {
      atomicExch(gen, it + 1);
    }
    while (atomicAdd(gen, 0) < it + 1) {
      __builtin_amdgcn_s_sleep(8);
    }
  }
  __syncthreads();
  __threadfence();     // acquire side
}

// ---------------- persistent LSTM recurrence ----------------
// grid = 64 WGs (one per 16-wide h-column tile), block = 256 (8 waves).
// Whh columns for this WG (4 gate tiles x 16 cols x K=1024) live in LDS for ALL steps.
// h_{t-1} (32x1024 bf16) is DMA'd into LDS each step via async-to-LDS; WMMA is ds-fed.
__global__ __launch_bounds__(256) void lstm_step_seq(
    const float* __restrict__ xw,      // (T, 32, 4096) f32 precomputed x@Wih + bih
    const __bf16* __restrict__ Whh,    // (1024, 4096) bf16
    const float* __restrict__ bhh,     // (4096)
    __bf16* __restrict__ Hbuf,         // (T+1, 32, 1024) bf16; block 0 zeros
    float* __restrict__ outF,          // (T, 32, 1024) f32 or nullptr
    float* __restrict__ hT,            // (32, 1024) f32 final h
    float* __restrict__ cT,            // (32, 1024) f32 final c
    int* bar_cnt, int* bar_gen, int nwg)
{
  __shared__ __bf16 lds_B[1024 * kLdsBStride];  // 144.0 KB: Whh columns, resident all steps
  __shared__ __bf16 lds_A[32 * kLdsAStride];    //  64.5 KB: h_{t-1} staged per step
  __shared__ float  lds_g[4][512];              //   8.0 KB: activated gate tiles
  __shared__ float  lds_c[512];                 //   2.0 KB: persistent cell state

  const int lane = threadIdx.x & 31;
  const int wave = threadIdx.x >> 5;
  const int mt = wave & 1;             // batch half (rows 16mt..16mt+15)
  const int gt = wave >> 1;            // 0=i,1=f,2=g,3=o
  const int wg = blockIdx.x;           // h-column tile (16 cols)

  // init cell state
  for (int i = threadIdx.x; i < 512; i += 256) lds_c[i] = 0.f;

  // one-time preload of Whh columns into LDS: 1024 rows x 4 gates x 16 cols
  for (int task = threadIdx.x; task < 4096; task += 256) {
    const int k = task >> 2;
    const int g = task & 3;
    const __bf16* src = Whh + (size_t)k * kG + g * kH + wg * 16;
    __bf16* dst = lds_B + (size_t)k * kLdsBStride + g * 16;
    *(v8bf*)(dst)     = *(const v8bf*)(src);
    *(v8bf*)(dst + 8) = *(const v8bf*)(src + 8);
  }
  __syncthreads();

  const v8f vzero = {0.f, 0.f, 0.f, 0.f, 0.f, 0.f, 0.f, 0.f};

  const int coll = lane & 15;
  const int rl = (lane >> 4) * 8;
  const int col = gt * kH + wg * 16 + coll;
  const float bias = bhh[col];

  for (int t = 0; t < kT; ++t) {
    // DMA h_{t-1} (32 x 1024 bf16 = 64KB) into LDS: async global->LDS, 16B per lane
    const __bf16* Aprev = Hbuf + (size_t)t * kBH;
#pragma unroll 4
    for (int task = threadIdx.x; task < 4096; task += 256) {
      const int row = task >> 7;
      const int ch = (task & 127) << 3;
      async_ld_lds_b128(lds_A + (size_t)row * kLdsAStride + ch,
                        Aprev + (size_t)row * kH + ch);
    }

    // prefetch this step's xw gate tile while the DMA is in flight
    const float* xwcol = xw + (size_t)t * kB4H + col;
#pragma unroll
    for (int r = 0; r < 8; ++r)
      __builtin_prefetch(xwcol + (size_t)(mt * 16 + rl + r) * kG, 0, 0);

    wait_asynccnt0();
    __syncthreads();

    // gate pre-activation: acc = h_{t-1} @ Whh[:, gate tile]
    v8f acc = vzero;
    for (int k0 = 0; k0 < kH; k0 += 32) {
      v16bf a = load_a_frag(lds_A, kLdsAStride, mt * 16, k0, lane);
      v16bf b = load_b_frag(lds_B, kLdsBStride, k0, gt * 16, lane);
      acc = bwmma(a, b, acc);
    }

#pragma unroll
    for (int r = 0; r < 8; ++r) {
      const int row = mt * 16 + rl + r;                 // batch index 0..31
      float v = acc[r] + xwcol[(size_t)row * kG] + bias;
      v = (gt == 2) ? tanhf(v) : 1.f / (1.f + __expf(-v));
      lds_g[gt][mt * 256 + (rl + r) * 16 + coll] = v;
    }
    __syncthreads();

    if (wave < 2) {                                     // waves 0,1 do the cell update
      const int m = wave;
#pragma unroll
      for (int r = 0; r < 8; ++r) {
        const int pos = r * 32 + lane;                  // 0..255
        const int idx = m * 256 + pos;
        const float iv = lds_g[0][idx];
        const float fv = lds_g[1][idx];
        const float gv = lds_g[2][idx];
        const float ov = lds_g[3][idx];
        const float cold = lds_c[idx];
        const float cn = fv * cold + iv * gv;
        const float hn = ov * tanhf(cn);
        lds_c[idx] = cn;
        const int brow = m * 16 + (pos >> 4);
        const int hcol = wg * 16 + (pos & 15);
        const size_t hidx = (size_t)brow * kH + hcol;
        Hbuf[(size_t)(t + 1) * kBH + hidx] = (__bf16)hn;
        if (outF) outF[(size_t)t * kBH + hidx] = hn;
        if (t == kT - 1) { hT[hidx] = hn; cT[hidx] = cn; }
      }
    }
    grid_barrier(bar_cnt, bar_gen, nwg, t);
  }
}

// ---------------- host launch ----------------
extern "C" void kernel_launch(void* const* d_in, const int* in_sizes, int n_in,
                              void* d_out, int out_size, void* d_ws, size_t ws_size,
                              hipStream_t stream) {
  (void)in_sizes; (void)n_in; (void)out_size; (void)ws_size;

  const float* X    = (const float*)d_in[0];
  const float* Wih0 = (const float*)d_in[1];
  const float* bih0 = (const float*)d_in[2];
  const float* Whh0 = (const float*)d_in[3];
  const float* bhh0 = (const float*)d_in[4];
  const float* Wih1 = (const float*)d_in[5];
  const float* bih1 = (const float*)d_in[6];
  const float* Whh1 = (const float*)d_in[7];
  const float* bhh1 = (const float*)d_in[8];

  float* out = (float*)d_out;                       // (T,B,H) then h_n (2,B,H) then c_n (2,B,H)
  float* hn  = out + (size_t)kT * kBH;              // h_n base
  float* cn  = hn + 2 * (size_t)kBH;                // c_n base

  char* w = (char*)d_ws;
  size_t o = 0;
  int* bar = (int*)(w + o);                 o += 256;
  float*  xw  = (float*)(w + o);            o += (size_t)kTB * kG * sizeof(float);      // 268 MB
  __bf16* Xbf = (__bf16*)(w + o);           o += (size_t)kTB * kCIN * sizeof(__bf16);   // 33.5 MB
  __bf16* W0i = (__bf16*)(w + o);           o += (size_t)kCIN * kG * sizeof(__bf16);
  __bf16* W0h = (__bf16*)(w + o);           o += (size_t)kH * kG * sizeof(__bf16);
  __bf16* W1i = (__bf16*)(w + o);           o += (size_t)kH * kG * sizeof(__bf16);
  __bf16* W1h = (__bf16*)(w + o);           o += (size_t)kH * kG * sizeof(__bf16);
  __bf16* H0  = (__bf16*)(w + o);           o += (size_t)(kT + 1) * kBH * sizeof(__bf16);
  __bf16* H1  = (__bf16*)(w + o);           o += (size_t)(kT + 1) * kBH * sizeof(__bf16);

  // init barrier + zero h_{-1} blocks (ws is poisoned, must clear what we read)
  zero_u32<<<1, 64, 0, stream>>>((unsigned*)bar, 2);
  zero_u32<<<64, 256, 0, stream>>>((unsigned*)H0, kBH * 2 / 4);
  zero_u32<<<64, 256, 0, stream>>>((unsigned*)H1, kBH * 2 / 4);

  // f32 -> bf16 conversions
  cvt_f32_bf16<<<512, 256, 0, stream>>>(X,    Xbf, (long)kTB * kCIN);
  cvt_f32_bf16<<<512, 256, 0, stream>>>(Wih0, W0i, (long)kCIN * kG);
  cvt_f32_bf16<<<512, 256, 0, stream>>>(Whh0, W0h, (long)kH * kG);
  cvt_f32_bf16<<<512, 256, 0, stream>>>(Wih1, W1i, (long)kH * kG);
  cvt_f32_bf16<<<512, 256, 0, stream>>>(Whh1, W1h, (long)kH * kG);

  dim3 ggrid(kTB / 64, kG / 256);

  // layer 0: xw = X @ Wih0 + bih0, then recurrence
  gemm_bias<<<ggrid, 256, 0, stream>>>(Xbf, W0i, bih0, xw, kTB, kCIN, kG);
  lstm_step_seq<<<64, 256, 0, stream>>>(xw, W0h, bhh0, H0, nullptr,
                                        hn, cn, bar, bar + 1, 64);

  // layer 1: xw = Hseq0 @ Wih1 + bih1 (reuse xw buffer), then recurrence -> d_out
  gemm_bias<<<ggrid, 256, 0, stream>>>(H0 + kBH, W1i, bih1, xw, kTB, kH, kG);
  zero_u32<<<1, 64, 0, stream>>>((unsigned*)bar, 2);
  lstm_step_seq<<<64, 256, 0, stream>>>(xw, W1h, bhh1, H1, out,
                                        hn + kBH, cn + kBH, bar, bar + 1, 64);
}
